// SelfAttention_11321533792330
// MI455X (gfx1250) — matrix operand order repaired and checked by reading, not compiled, not run on hardware
//
#include <hip/hip_runtime.h>
#include <hip/hip_bf16.h>

// ---------------------------------------------------------------------------
// MI455X (gfx1250) self-attention, f16 WMMA path.
// B=8, S=2048, D=256.  ~41 GFLOP vs ~18 MB HBM traffic -> compute bound,
// so everything runs through v_wmma_f32_16x16x32_f16 (f32 accumulate).
// Workspace: Qg[BS][D] f16 (8MB) | Kg[BS][D] f16 (8MB) | Vt[B*D][S] f16 (8MB)
//            | Wh[3][D*D] f16 (384KB, pre-converted weights, L2-resident)
// ---------------------------------------------------------------------------

typedef __attribute__((ext_vector_type(16))) _Float16 v16h;
typedef __attribute__((ext_vector_type(8)))  float    v8f;

struct __align__(16) U4 { unsigned int a, b, c, d; };
struct __align__(8)  U2 { unsigned int a, b; };
struct __align__(16) F4 { float x, y, z, w; };

union AFrag {                // 16x32 f16 A-fragment OR 32x16 f16 B-fragment
    v16h      v;
    _Float16  h[16];
    U4        u[2];
};
union CFrag {                // 16x16 f32 C/D fragment
    v8f   v;
    float f[8];
};
union H4 {                   // 4 halves <-> 8-byte store
    _Float16 h[4];
    U2       u;
};

#define Dh 256
#define Sh 2048
#define Bh 8

static __device__ __forceinline__ v8f wmma_f16(v16h a, v16h b, v8f c) {
    // D = A(16x32 f16) * B(32x16 f16) + C(16x16 f32)
    return __builtin_amdgcn_wmma_f32_16x16x32_f16(
        /*neg_a=*/false, a, /*neg_b=*/false, b,
        /*c_mod=*/(short)0, c, /*reuse_a=*/false, /*reuse_b=*/false);
}

// float atomic-max via monotonic int trick (works with -inf init bits 0xFF800000)
static __device__ __forceinline__ void atomicMaxFloat(float* addr, float v) {
    if (v >= 0.0f) atomicMax((int*)addr,          __float_as_int(v));
    else           atomicMin((unsigned int*)addr, __float_as_uint(v));
}

// ---------------------------------------------------------------------------
__global__ void init_out_kernel(unsigned int* o, int n) {
    int i = blockIdx.x * blockDim.x + threadIdx.x;
    if (i < n) o[i] = 0xFF800000u;   // -inf
}

// ---------------------------------------------------------------------------
// One-shot fp32 -> f16 conversion of the three weight matrices (row-major
// preserved).  3*64K elements; done once so the hot GEMM loads f16 directly.
// ---------------------------------------------------------------------------
__global__ void cvt_w_kernel(const float* __restrict__ Wq,
                             const float* __restrict__ Wk,
                             const float* __restrict__ Wv,
                             _Float16* __restrict__ Wh)
{
    const int quads_per_mat = (Dh * Dh) / 4;              // 16384
    int i = blockIdx.x * blockDim.x + threadIdx.x;        // quad index
    if (i >= 3 * quads_per_mat) return;
    const int m = i / quads_per_mat;
    const int q = i - m * quads_per_mat;
    const float* src = (m == 0) ? Wq : (m == 1) ? Wk : Wv;
    const F4 f = *(const F4*)(src + (size_t)q * 4);
    H4 h;
    h.h[0] = (_Float16)f.x; h.h[1] = (_Float16)f.y;
    h.h[2] = (_Float16)f.z; h.h[3] = (_Float16)f.w;
    *(U2*)(Wh + (size_t)i * 4) = h.u;
}

// ---------------------------------------------------------------------------
// QKV projection: Y[row][e] = sum_d x[row][d] * W[e][d] + b[e]
// grid = (B*S)/16 = 1024 blocks of 128 threads (4 waves).
// Wave w owns output columns [w*64, w*64+64) for all three matrices.
// ---------------------------------------------------------------------------
__global__ __launch_bounds__(128) void qkv_proj_kernel(
    const float* __restrict__ x,
    const _Float16* __restrict__ Wh,     // [3][D*D] f16, row-major like W
    const float* __restrict__ bq, const float* __restrict__ bk,
    const float* __restrict__ bv,
    _Float16* __restrict__ Qg, _Float16* __restrict__ Kg,
    _Float16* __restrict__ Vt)
{
    const int tid  = threadIdx.x;
    const int wv   = tid >> 5;
    const int lane = tid & 31;
    const int g    = lane >> 4;      // half-wave group
    const int ln   = lane & 15;
    const int row0 = blockIdx.x * 16;

    // ---- load + convert the 16x256 x row-tile into 8 A-fragments ----------
    AFrag xa[8];
    const float* xr = x + (size_t)(row0 + ln) * Dh;   // A: lane holds row M=ln
#pragma unroll
    for (int c = 0; c < 8; ++c) {
        const int d0 = 32 * c + 8 * g;                // halves 0..7 -> K=d0+0..7
        const F4 f0 = *(const F4*)(xr + d0);
        const F4 f1 = *(const F4*)(xr + d0 + 4);
        const F4 f2 = *(const F4*)(xr + d0 + 16);     // halves 8..15 -> K=d0+16..23
        const F4 f3 = *(const F4*)(xr + d0 + 20);
        float tf[16] = { f0.x,f0.y,f0.z,f0.w, f1.x,f1.y,f1.z,f1.w,
                         f2.x,f2.y,f2.z,f2.w, f3.x,f3.y,f3.z,f3.w };
#pragma unroll
        for (int i = 0; i < 16; ++i) xa[c].h[i] = (_Float16)tf[i];
    }

    const float* bs[3] = { bq, bk, bv };

#pragma unroll
    for (int m = 0; m < 3; ++m) {
        const _Float16* Wm = Wh + (size_t)m * Dh * Dh;
#pragma unroll
        for (int t = 0; t < 4; ++t) {
            const int ncol0 = wv * 64 + t * 16;
            CFrag acc;
#pragma unroll
            for (int r = 0; r < 8; ++r) acc.f[r] = 0.0f;

            // B-fragment: lane holds W[ncol0+ln][32c+8g .. +7] and [32c+8g+16 .. +23]
            const _Float16* wr = Wm + (size_t)(ncol0 + ln) * Dh + 8 * g;
#pragma unroll
            for (int c = 0; c < 8; ++c) {
                AFrag bt;
                bt.u[0] = *(const U4*)(wr + 32 * c);
                bt.u[1] = *(const U4*)(wr + 32 * c + 16);
                acc.v = wmma_f16(xa[c].v, bt.v, acc.v);
            }

            const float bb = bs[m][ncol0 + ln];
#pragma unroll
            for (int r = 0; r < 8; ++r) acc.f[r] += bb;

            if (m < 2) {
                // Q / K row-major f16: element (r,lane) -> row row0+r+8g, col ncol0+ln
                _Float16* dst = (m == 0) ? Qg : Kg;
#pragma unroll
                for (int r = 0; r < 8; ++r)
                    dst[(size_t)(row0 + r + 8 * g) * Dh + ncol0 + ln] = (_Float16)acc.f[r];
            } else {
                // V transposed: Vt[(b*D + e)*S + s]; consecutive r -> consecutive s
                const int b  = row0 >> 11;               // 2048 rows per batch
                const int s0 = (row0 & 2047) + 8 * g;
                const int e  = ncol0 + ln;
                AFrag pk;
#pragma unroll
                for (int r = 0; r < 8; ++r) pk.h[r] = (_Float16)acc.f[r];
                *(U4*)(Vt + ((size_t)(b * Dh + e)) * Sh + s0) = pk.u[0];
            }
        }
    }
}

// ---------------------------------------------------------------------------
// Flash attention + row-max pooling.
// grid = B * (S/16) = 1024 blocks of 128 threads (4 waves).
// Per block: one (batch b, 16-query tile). Wave w computes score columns
// [kb+16w, kb+16w+16) each iteration and owns D columns [64w, 64w+64) of O.
// ---------------------------------------------------------------------------
__global__ __launch_bounds__(128) void attn_kernel(
    const _Float16* __restrict__ Qg, const _Float16* __restrict__ Kg,
    const _Float16* __restrict__ Vt, float* __restrict__ out)
{
    __shared__ __align__(16) _Float16 Pl[16 * 64];   // probability tile, f16
    __shared__ __align__(16) float    pmax[16 * 4];  // per-row max partials [row][wave]
    __shared__ __align__(16) float    psum[16 * 4];  // per-row sum partials [row][wave]

    const int tid  = threadIdx.x;
    const int wv   = tid >> 5;
    const int lane = tid & 31;
    const int g    = lane >> 4;
    const int ln   = lane & 15;
    const int b    = blockIdx.x >> 7;     // 128 query tiles per batch
    const int qt   = blockIdx.x & 127;

    // ---- Q tile (16x256 f16) as 8 cached A-fragments -----------------------
    AFrag qa[8];
    const _Float16* qrow = Qg + (size_t)(b * Sh + qt * 16 + ln) * Dh;
#pragma unroll
    for (int c = 0; c < 8; ++c) {
        const int d0 = 32 * c + 8 * g;
        qa[c].u[0] = *(const U4*)(qrow + d0);         // K = d0+0..7
        qa[c].u[1] = *(const U4*)(qrow + d0 + 16);    // K = d0+16..23
    }

    float m_run[8], lsum[8];
#pragma unroll
    for (int r = 0; r < 8; ++r) { m_run[r] = -1e30f; lsum[r] = 0.0f; }
    CFrag o[4];
#pragma unroll
    for (int t = 0; t < 4; ++t)
#pragma unroll
        for (int r = 0; r < 8; ++r) o[t].f[r] = 0.0f;

    for (int kb = 0; kb < Sh; kb += 64) {
        // ---- 1. score tile S = Q * K^T for this wave's 16 k-columns --------
        CFrag s;
#pragma unroll
        for (int r = 0; r < 8; ++r) s.f[r] = 0.0f;
        const _Float16* krow = Kg + (size_t)(b * Sh + kb + wv * 16 + ln) * Dh + 8 * g;
#pragma unroll
        for (int c = 0; c < 8; ++c) {
            AFrag kt;   // B-fragment: lane holds K[col][32c+8g .. +7] and [32c+8g+16 .. +23]
            kt.u[0] = *(const U4*)(krow + 32 * c);
            kt.u[1] = *(const U4*)(krow + 32 * c + 16);
            s.v = wmma_f16(qa[c].v, kt.v, s.v);
        }

        // ---- 2. per-row max partial (16-lane reduce), publish to LDS -------
        float tm[8];
#pragma unroll
        for (int r = 0; r < 8; ++r) tm[r] = s.f[r];
#pragma unroll
        for (int msk = 1; msk < 16; msk <<= 1)
#pragma unroll
            for (int r = 0; r < 8; ++r)
                tm[r] = fmaxf(tm[r], __shfl_xor(tm[r], msk, 32));
        if (ln == 0) {
#pragma unroll
            for (int r = 0; r < 8; ++r) pmax[(8 * g + r) * 4 + wv] = tm[r];
        }
        __syncthreads();

        // ---- 3. online-softmax rescale + exponentiate ----------------------
        float p[8];
#pragma unroll
        for (int r = 0; r < 8; ++r) {
            const F4 pm = ((const F4*)pmax)[r + 8 * g];
            const float tmax = fmaxf(fmaxf(pm.x, pm.y), fmaxf(pm.z, pm.w));
            const float mnew = fmaxf(m_run[r], tmax);
            const float sc   = __expf(m_run[r] - mnew);
            lsum[r] *= sc;
#pragma unroll
            for (int t = 0; t < 4; ++t) o[t].f[r] *= sc;
            p[r]     = __expf(s.f[r] - mnew);
            m_run[r] = mnew;
        }

        // ---- 4. row-sum partials + share P (f16) through LDS ---------------
        float ts[8];
#pragma unroll
        for (int r = 0; r < 8; ++r) ts[r] = p[r];
#pragma unroll
        for (int msk = 1; msk < 16; msk <<= 1)
#pragma unroll
            for (int r = 0; r < 8; ++r) ts[r] += __shfl_xor(ts[r], msk, 32);
        if (ln == 0) {
#pragma unroll
            for (int r = 0; r < 8; ++r) psum[(8 * g + r) * 4 + wv] = ts[r];
        }
#pragma unroll
        for (int r = 0; r < 8; ++r)
            Pl[(r + 8 * g) * 64 + wv * 16 + ln] = (_Float16)p[r];
        __syncthreads();

#pragma unroll
        for (int r = 0; r < 8; ++r) {
            const F4 ps = ((const F4*)psum)[r + 8 * g];
            lsum[r] += ps.x + ps.y + ps.z + ps.w;
        }

        // ---- 5. O += P * V on this wave's 64-column D slice ----------------
#pragma unroll
        for (int c = 0; c < 2; ++c) {
            AFrag pa;   // A-fragment of P from LDS (row M=ln)
            const _Float16* pb = Pl + ln * 64 + 32 * c + 8 * g;
            pa.u[0] = *(const U4*)(pb);
            pa.u[1] = *(const U4*)(pb + 16);
#pragma unroll
            for (int t = 0; t < 4; ++t) {
                const int n = wv * 64 + t * 16 + ln;
                const _Float16* vb = Vt + ((size_t)(b * Dh + n)) * Sh + kb + 32 * c + 8 * g;
                AFrag vt;   // B-fragment of V (contiguous thanks to Vt transpose)
                vt.u[0] = *(const U4*)(vb);
                vt.u[1] = *(const U4*)(vb + 16);
                o[t].v = wmma_f16(pa.v, vt.v, o[t].v);
            }
        }
    }

    // ---- epilogue: normalize (8 reciprocals, not 32 divides), max over the
    // 16 query rows, then global atomic-max into out[b][col] ----------------
    float inv[8];
#pragma unroll
    for (int r = 0; r < 8; ++r) inv[r] = 1.0f / lsum[r];
#pragma unroll
    for (int t = 0; t < 4; ++t) {
        float cm = -1e30f;
#pragma unroll
        for (int r = 0; r < 8; ++r) cm = fmaxf(cm, o[t].f[r] * inv[r]);
        cm = fmaxf(cm, __shfl_xor(cm, 16, 32));   // combine rows 0-7 with 8-15
        if (lane < 16)
            atomicMaxFloat(out + b * Dh + wv * 64 + t * 16 + ln, cm);
    }
}

// ---------------------------------------------------------------------------
extern "C" void kernel_launch(void* const* d_in, const int* in_sizes, int n_in,
                              void* d_out, int out_size, void* d_ws, size_t ws_size,
                              hipStream_t stream) {
    (void)in_sizes; (void)n_in; (void)ws_size;
    const float* x  = (const float*)d_in[0];
    const float* Wq = (const float*)d_in[1];
    const float* bq = (const float*)d_in[2];
    const float* Wk = (const float*)d_in[3];
    const float* bk = (const float*)d_in[4];
    const float* Wv = (const float*)d_in[5];
    const float* bv = (const float*)d_in[6];

    _Float16* Qg = (_Float16*)d_ws;                       // 8 MB
    _Float16* Kg = Qg + (size_t)Bh * Sh * Dh;             // 8 MB
    _Float16* Vt = Kg + (size_t)Bh * Sh * Dh;             // 8 MB (transposed V)
    _Float16* Wh = Vt + (size_t)Bh * Sh * Dh;             // 384 KB (f16 weights)

    init_out_kernel<<<(out_size + 255) / 256, 256, 0, stream>>>(
        (unsigned int*)d_out, out_size);

    cvt_w_kernel<<<(3 * (Dh * Dh / 4) + 255) / 256, 256, 0, stream>>>(
        Wq, Wk, Wv, Wh);

    qkv_proj_kernel<<<(Bh * Sh) / 16, 128, 0, stream>>>(
        x, Wh, bq, bk, bv, Qg, Kg, Vt);

    attn_kernel<<<Bh * (Sh / 16), 128, 0, stream>>>(
        Qg, Kg, Vt, (float*)d_out);
}
